// MoELayer_88424786690743
// MI455X (gfx1250) — matrix operand order repaired and checked by reading, not compiled, not run on hardware
//
#include <hip/hip_runtime.h>
#include <hip/hip_bf16.h>

// ---------------- problem dims (fixed by reference) ----------------
#define T_TOK 8192   // B*S = 4*2048
#define DDIM  1024
#define HDIM  2048
#define NE    8

typedef __attribute__((ext_vector_type(16))) __bf16 v16bf;
typedef __attribute__((ext_vector_type(8)))  float  v8f;
typedef __attribute__((ext_vector_type(4)))  unsigned int v4u;
typedef __attribute__((ext_vector_type(8)))  int v8i_;
typedef __attribute__((ext_vector_type(4)))  int v4i;

// ---------------- TDM availability ----------------
#if defined(__has_builtin)
#if __has_builtin(__builtin_amdgcn_tensor_load_to_lds) && __has_builtin(__builtin_amdgcn_s_wait_tensorcnt)
#define USE_TDM 1
#endif
#endif
#ifndef USE_TDM
#define USE_TDM 0
#endif

#if USE_TDM
#if __has_include(<hip/amd_detail/amd_gfx1250_TDM.h>)
// amdgpu-toolchain (therock) 6-arg form
#define TDM_LOAD(g0, g1, g2, g3) \
    do { v8i_ _z = {}; __builtin_amdgcn_tensor_load_to_lds((g0), (g1), (g2), (g3), _z, 0); } while (0)
#else
// ROCm 7.2 5-arg form
#define TDM_LOAD(g0, g1, g2, g3) __builtin_amdgcn_tensor_load_to_lds((g0), (g1), (g2), (g3), 0)
#endif
#define TDM_WAIT() __builtin_amdgcn_s_wait_tensorcnt(0)
#else
#define TDM_WAIT() do {} while (0)
#endif

union FragU { v16bf v; unsigned u[8]; };

// fp32 -> bf16 (round to nearest even), raw bits
__device__ inline unsigned short f2bf(float f) {
    union { float f; unsigned u; } cv;
    cv.f = f;
    unsigned u = cv.u;
    unsigned r = u + 0x7fffu + ((u >> 16) & 1u);
    return (unsigned short)(r >> 16);
}

// A-matrix 16x32 bf16 fragment (ISA 7.12.2): lane m = lane&15, kh = lane>>4
// uint pair j=0..3 -> K = 2j + 8*kh ; j=4..7 -> K = 16 + 2(j-4) + 8*kh
__device__ inline v16bf load_fragA(const unsigned short* rowp, int kh) {
    FragU f;
    const unsigned* p = (const unsigned*)rowp;
    int b = kh * 4;
#pragma unroll
    for (int j = 0; j < 4; ++j) {
        f.u[j]     = p[b + j];
        f.u[4 + j] = p[8 + b + j];
    }
    return f.v;
}

// B-matrix 32x16 bf16 fragment: lane n = lane&15, kh = lane>>4; uint j=0..3 -> K = 2j + 8*kh ; j=4..7 -> K = 16 + 2(j-4) + 8*kh
__device__ inline v16bf load_fragB(const unsigned short* rowp, int kh) {
    FragU f;
    const unsigned* p = (const unsigned*)rowp;
    int b = kh * 4;
#pragma unroll
    for (int j = 0; j < 4; ++j) {
        f.u[j]     = p[b + j];
        f.u[4 + j] = p[8 + b + j];
    }
    return f.v;
}

// ---------------- simple kernels ----------------

__global__ void k_init(float* __restrict__ out, int n, int* __restrict__ counts) {
    int i = blockIdx.x * blockDim.x + threadIdx.x;
    if (i < n) out[i] = 0.0f;
    if (i < NE) counts[i] = 0;
}

__global__ void k_cvt(const float* __restrict__ s, unsigned short* __restrict__ d, int n) {
    int i = blockIdx.x * blockDim.x + threadIdx.x;
    if (i < n) d[i] = f2bf(s[i]);
}

// one wave32 per token: router logits, softmax, top-2, renorm, append to lists
__global__ __launch_bounds__(256) void k_router(
    const float* __restrict__ x, const float* __restrict__ rw,
    float* __restrict__ combine, int* __restrict__ counts, int* __restrict__ lists)
{
    int wave = threadIdx.x >> 5;
    int lane = threadIdx.x & 31;
    int t = blockIdx.x * 8 + wave;
    if (t >= T_TOK) return;
    const float* xr = x + (size_t)t * DDIM;

    float acc[NE];
#pragma unroll
    for (int e = 0; e < NE; ++e) acc[e] = 0.0f;
    for (int d = lane; d < DDIM; d += 32) {
        float xv = xr[d];
#pragma unroll
        for (int e = 0; e < NE; ++e) acc[e] += xv * rw[e * DDIM + d];
    }
#pragma unroll
    for (int e = 0; e < NE; ++e) {
#pragma unroll
        for (int off = 16; off > 0; off >>= 1)
            acc[e] += __shfl_xor(acc[e], off, 32);
    }

    if (lane == 0) {
        float mx = acc[0];
#pragma unroll
        for (int e = 1; e < NE; ++e) mx = fmaxf(mx, acc[e]);
        float p[NE];
#pragma unroll
        for (int e = 0; e < NE; ++e) p[e] = __expf(acc[e] - mx);
        int i1 = 0; float b1 = p[0];
#pragma unroll
        for (int e = 1; e < NE; ++e) if (p[e] > b1) { b1 = p[e]; i1 = e; }
        int i2 = (i1 == 0) ? 1 : 0; float b2 = p[i2];
#pragma unroll
        for (int e = 0; e < NE; ++e)
            if (e != i1 && p[e] > b2) { b2 = p[e]; i2 = e; }
        float inv = 1.0f / (b1 + b2);
#pragma unroll
        for (int e = 0; e < NE; ++e) combine[(size_t)t * NE + e] = 0.0f;
        combine[(size_t)t * NE + i1] = b1 * inv;
        combine[(size_t)t * NE + i2] = b2 * inv;
        int s1 = atomicAdd(&counts[i1], 1);
        lists[(size_t)i1 * T_TOK + s1] = t;
        int s2 = atomicAdd(&counts[i2], 1);
        lists[(size_t)i2 * T_TOK + s2] = t;
    }
}

// ---------------- fused gate/up GEMM ----------------
// Block = 8 waves as 2(row)x4(col); block tile = 64 tokens x 128 h.
// Wave tile = 32 tokens x 32 h for BOTH gate and up (8 accumulators).
// A-tiles (gathered token rows) staged in LDS via TDM gather-mode, double buffered.
__global__ __launch_bounds__(256) void k_upgate(
    const unsigned short* __restrict__ xb,
    const unsigned short* __restrict__ gb,   // pre-offset to expert e
    const unsigned short* __restrict__ ub,   // pre-offset to expert e
    const int* __restrict__ lists,
    const int* __restrict__ counts,
    unsigned short* __restrict__ act,
    int e)
{
    int cnt = counts[e];
    int tile_r = blockIdx.x * 64;
    if (tile_r >= cnt) return;

    __shared__ __align__(16) unsigned short Abuf[2][64][32];

    int tid  = threadIdx.x;
    int wave = tid >> 5;
    int lane = tid & 31;
    int wrow = wave >> 2;      // 0..1 -> 32-token sub-tile
    int wcol = wave & 3;       // 0..3 -> 32-h sub-tile
    int m  = lane & 15;
    int kh = lane >> 4;

    const int KT = DDIM / 32;

#if USE_TDM
    // ---- constant descriptor group1: 2B elements, gather 2D tile 32 x 16 rows ----
    v8i_ g1;
    g1[0] = (1 << 16);                        // data_size = 2 bytes
    g1[1] = (DDIM & 0xffff) << 16;            // tensor_dim0 lo16
    g1[2] = (T_TOK & 0xffff) << 16;           // dim0 hi=0 | tensor_dim1 lo16
    g1[3] = (32 << 16);                       // dim1 hi=0 | tile_dim0 = 32
    g1[4] = 16;                               // tile_dim1 = #indices = 16
    g1[5] = DDIM;                             // tensor_dim0_stride lo32
    g1[6] = 0;
    g1[7] = 0;
    // ---- per-wave row-index groups 2/3 (waves 0..3 own 16-row groups) ----
    int ridx = tile_r + wave * 16 + (lane & 15);
    int rcl  = (ridx < cnt) ? ridx : (cnt - 1);
    int idx  = lists[(size_t)e * T_TOK + rcl] & 0xffff;
    int pk[8];
#pragma unroll
    for (int j = 0; j < 8; ++j)
        pk[j] = (__builtin_amdgcn_readlane(idx, 2 * j) & 0xffff) |
                ((__builtin_amdgcn_readlane(idx, 2 * j + 1) & 0xffff) << 16);
    v4i g2 = { pk[0], pk[1], pk[2], pk[3] };
    v4i g3 = { pk[4], pk[5], pk[6], pk[7] };
    unsigned abase = (unsigned)(size_t)&Abuf[0][0][0];
    unsigned long long gbase = (unsigned long long)(size_t)xb;

    if (wave < 4) {  // prologue: tile k=0 into buf 0
        v4u g0;
        g0[0] = 1u | (1u << 31);                       // count=1, gather_mode
        g0[1] = abase + (unsigned)(wave * 16 * 64);    // lds byte addr (row=64B)
        unsigned long long ga = gbase;                 // k = 0
        g0[2] = (unsigned)ga;
        g0[3] = ((unsigned)(ga >> 32) & 0x1FFFFFFu) | (2u << 30);
        TDM_LOAD(g0, g1, g2, g3);
    }
#else
    {   // prologue copy, k = 0: 256 threads, 16B each; 64 rows x 64B
        int row = tid >> 2, seg = tid & 3;
        int rr = tile_r + row;
        int rc = (rr < cnt) ? rr : (cnt - 1);
        int tok = lists[(size_t)e * T_TOK + rc];
        *(uint4*)&Abuf[0][row][seg * 8] = *(const uint4*)&xb[(size_t)tok * DDIM + seg * 8];
    }
#endif

    // B row pointers: gate/up rows h, contiguous K
    int h0 = blockIdx.y * 128 + wcol * 32 + m;
    const unsigned short* gr0 = gb + (size_t)h0 * DDIM;
    const unsigned short* gr1 = gb + (size_t)(h0 + 16) * DDIM;
    const unsigned short* ur0 = ub + (size_t)h0 * DDIM;
    const unsigned short* ur1 = ub + (size_t)(h0 + 16) * DDIM;

    v8f accg[2][2] = {};
    v8f accu[2][2] = {};

    for (int kt = 0; kt < KT; ++kt) {
        int cur = kt & 1;
        TDM_WAIT();
        __syncthreads();
        if (kt + 1 < KT) {
            int nk = (kt + 1) * 32;
#if USE_TDM
            if (wave < 4) {
                v4u g0;
                g0[0] = 1u | (1u << 31);
                g0[1] = abase + (unsigned)((cur ^ 1) * 4096 + wave * 16 * 64);
                unsigned long long ga = gbase + (unsigned long long)(2 * nk);
                g0[2] = (unsigned)ga;
                g0[3] = ((unsigned)(ga >> 32) & 0x1FFFFFFu) | (2u << 30);
                TDM_LOAD(g0, g1, g2, g3);
            }
#else
            {
                int row = tid >> 2, seg = tid & 3;
                int rr = tile_r + row;
                int rc = (rr < cnt) ? rr : (cnt - 1);
                int tok = lists[(size_t)e * T_TOK + rc];
                *(uint4*)&Abuf[cur ^ 1][row][seg * 8] =
                    *(const uint4*)&xb[(size_t)tok * DDIM + nk + seg * 8];
            }
#endif
        }
        int k = kt * 32;
        v16bf a0 = load_fragA(&Abuf[cur][wrow * 32 + m][0], kh);
        v16bf a1 = load_fragA(&Abuf[cur][wrow * 32 + 16 + m][0], kh);
        v16bf bg0 = load_fragB(gr0 + k, kh);
        v16bf bg1 = load_fragB(gr1 + k, kh);
        v16bf bu0 = load_fragB(ur0 + k, kh);
        v16bf bu1 = load_fragB(ur1 + k, kh);
        accg[0][0] = __builtin_amdgcn_wmma_f32_16x16x32_bf16(false, a0, false, bg0, (short)0, accg[0][0], false, false);
        accg[0][1] = __builtin_amdgcn_wmma_f32_16x16x32_bf16(false, a0, false, bg1, (short)0, accg[0][1], false, false);
        accg[1][0] = __builtin_amdgcn_wmma_f32_16x16x32_bf16(false, a1, false, bg0, (short)0, accg[1][0], false, false);
        accg[1][1] = __builtin_amdgcn_wmma_f32_16x16x32_bf16(false, a1, false, bg1, (short)0, accg[1][1], false, false);
        accu[0][0] = __builtin_amdgcn_wmma_f32_16x16x32_bf16(false, a0, false, bu0, (short)0, accu[0][0], false, false);
        accu[0][1] = __builtin_amdgcn_wmma_f32_16x16x32_bf16(false, a0, false, bu1, (short)0, accu[0][1], false, false);
        accu[1][0] = __builtin_amdgcn_wmma_f32_16x16x32_bf16(false, a1, false, bu0, (short)0, accu[1][0], false, false);
        accu[1][1] = __builtin_amdgcn_wmma_f32_16x16x32_bf16(false, a1, false, bu1, (short)0, accu[1][1], false, false);
        __syncthreads();
    }

    int hc0 = blockIdx.y * 128 + wcol * 32 + (lane & 15);
#pragma unroll
    for (int i = 0; i < 2; ++i) {
#pragma unroll
        for (int r = 0; r < 8; ++r) {
            int rr = tile_r + wrow * 32 + i * 16 + r + kh * 8;
            if (rr < cnt) {
#pragma unroll
                for (int j = 0; j < 2; ++j) {
                    float g = accg[i][j][r];
                    float u = accu[i][j][r];
                    float s = g / (1.0f + __expf(-g));
                    act[(size_t)rr * HDIM + hc0 + j * 16] = f2bf(s * u);
                }
            }
        }
    }
}

// ---------------- down GEMM + weighted scatter ----------------
// Block = 8 waves as 2x4; block tile = 64 rows x 256 d. Wave tile = 32 x 64 (8 accums).
// A-tile (contiguous act rows) staged via TDM normal-mode 2D descriptor, double buffered.
__global__ __launch_bounds__(256) void k_down(
    const unsigned short* __restrict__ act,
    const unsigned short* __restrict__ db,   // pre-offset to expert e
    const int* __restrict__ lists,
    const int* __restrict__ counts,
    const float* __restrict__ combine,
    float* __restrict__ out,
    int e)
{
    int cnt = counts[e];
    int tile_r = blockIdx.x * 64;
    if (tile_r >= cnt) return;

    __shared__ __align__(16) unsigned short Abuf[2][64][32];

    int tid  = threadIdx.x;
    int wave = tid >> 5;
    int lane = tid & 31;
    int wrow = wave >> 2;
    int wcol = wave & 3;
    int m  = lane & 15;
    int kh = lane >> 4;

    const int KT = HDIM / 32;

#if USE_TDM
    v8i_ g1;
    g1[0] = (1 << 16);                        // data_size = 2B
    g1[1] = (HDIM & 0xffff) << 16;            // tensor_dim0 lo16
    g1[2] = (T_TOK & 0xffff) << 16;           // tensor_dim1 lo16
    g1[3] = (32 << 16);                       // tile_dim0 = 32
    g1[4] = 64;                               // tile_dim1 = 64 rows
    g1[5] = HDIM;                             // tensor_dim0_stride
    g1[6] = 0;
    g1[7] = 0;
    v4i g2 = {};
    v4i g3 = {};
    unsigned abase = (unsigned)(size_t)&Abuf[0][0][0];
    unsigned long long gbase = (unsigned long long)(size_t)(act) +
                               (unsigned long long)2 * ((size_t)tile_r * HDIM);
    if (wave == 0) {
        v4u g0;
        g0[0] = 1u;
        g0[1] = abase;
        unsigned long long ga = gbase;
        g0[2] = (unsigned)ga;
        g0[3] = ((unsigned)(ga >> 32) & 0x1FFFFFFu) | (2u << 30);
        TDM_LOAD(g0, g1, g2, g3);
    }
#else
    {
        int row = tid >> 2, seg = tid & 3;
        *(uint4*)&Abuf[0][row][seg * 8] =
            *(const uint4*)&act[((size_t)(tile_r + row)) * HDIM + seg * 8];
    }
#endif

    int d0 = blockIdx.y * 256 + wcol * 64 + m;
    const unsigned short* br[4];
#pragma unroll
    for (int j = 0; j < 4; ++j) br[j] = db + (size_t)(d0 + j * 16) * HDIM;

    v8f acc[2][4] = {};

    for (int kt = 0; kt < KT; ++kt) {
        int cur = kt & 1;
        TDM_WAIT();
        __syncthreads();
        if (kt + 1 < KT) {
            int nk = (kt + 1) * 32;
#if USE_TDM
            if (wave == 0) {
                v4u g0;
                g0[0] = 1u;
                g0[1] = abase + (unsigned)((cur ^ 1) * 4096);
                unsigned long long ga = gbase + (unsigned long long)(2 * nk);
                g0[2] = (unsigned)ga;
                g0[3] = ((unsigned)(ga >> 32) & 0x1FFFFFFu) | (2u << 30);
                TDM_LOAD(g0, g1, g2, g3);
            }
#else
            {
                int row = tid >> 2, seg = tid & 3;
                *(uint4*)&Abuf[cur ^ 1][row][seg * 8] =
                    *(const uint4*)&act[((size_t)(tile_r + row)) * HDIM + nk + seg * 8];
            }
#endif
        }
        int k = kt * 32;
        v16bf a0 = load_fragA(&Abuf[cur][wrow * 32 + m][0], kh);
        v16bf a1 = load_fragA(&Abuf[cur][wrow * 32 + 16 + m][0], kh);
#pragma unroll
        for (int j = 0; j < 4; ++j) {
            v16bf b = load_fragB(br[j] + k, kh);
            acc[0][j] = __builtin_amdgcn_wmma_f32_16x16x32_bf16(false, a0, false, b, (short)0, acc[0][j], false, false);
            acc[1][j] = __builtin_amdgcn_wmma_f32_16x16x32_bf16(false, a1, false, b, (short)0, acc[1][j], false, false);
        }
        __syncthreads();
    }

    int dc0 = blockIdx.y * 256 + wcol * 64 + (lane & 15);
#pragma unroll
    for (int i = 0; i < 2; ++i) {
#pragma unroll
        for (int r = 0; r < 8; ++r) {
            int rr = tile_r + wrow * 32 + i * 16 + r + kh * 8;
            if (rr < cnt) {
                int t = lists[(size_t)e * T_TOK + rr];
                float w = combine[(size_t)t * NE + e];
#pragma unroll
                for (int j = 0; j < 4; ++j) {
                    size_t o = (size_t)t * DDIM + dc0 + j * 16;
                    out[o] += w * acc[i][j][r];
                }
            }
        }
    }
}

// ---------------- launcher ----------------

extern "C" void kernel_launch(void* const* d_in, const int* in_sizes, int n_in,
                              void* d_out, int out_size, void* d_ws, size_t ws_size,
                              hipStream_t stream) {
    const float* x  = (const float*)d_in[0];
    const float* rw = (const float*)d_in[1];
    const float* gw = (const float*)d_in[2];
    const float* uw = (const float*)d_in[3];
    const float* dw = (const float*)d_in[4];
    float* out = (float*)d_out;

    char* ws = (char*)d_ws;
    size_t off = 0;
    float* combine = (float*)(ws + off); off += (size_t)T_TOK * NE * 4;
    int*   counts  = (int*)(ws + off);   off += 256;
    int*   lists   = (int*)(ws + off);   off += (size_t)NE * T_TOK * 4;
    unsigned short* xb  = (unsigned short*)(ws + off); off += (size_t)T_TOK * DDIM * 2;
    unsigned short* gb  = (unsigned short*)(ws + off); off += (size_t)NE * HDIM * DDIM * 2;
    unsigned short* ub  = (unsigned short*)(ws + off); off += (size_t)NE * HDIM * DDIM * 2;
    unsigned short* db  = (unsigned short*)(ws + off); off += (size_t)NE * DDIM * HDIM * 2;
    unsigned short* act = (unsigned short*)(ws + off); off += (size_t)T_TOK * HDIM * 2;

    const int nOut = T_TOK * DDIM;
    k_init<<<(nOut + 255) / 256, 256, 0, stream>>>(out, nOut, counts);
    k_router<<<T_TOK / 8, 256, 0, stream>>>(x, rw, combine, counts, lists);

    k_cvt<<<(nOut + 255) / 256, 256, 0, stream>>>(x, xb, nOut);
    const int nW = NE * HDIM * DDIM;
    k_cvt<<<(nW + 255) / 256, 256, 0, stream>>>(gw, gb, nW);
    k_cvt<<<(nW + 255) / 256, 256, 0, stream>>>(uw, ub, nW);
    k_cvt<<<(nW + 255) / 256, 256, 0, stream>>>(dw, db, nW);

    for (int e = 0; e < NE; ++e) {
        dim3 g1(T_TOK / 64, HDIM / 128);  // 128 x 16, early exit past counts[e]
        k_upgate<<<g1, 256, 0, stream>>>(xb,
                                         gb + (size_t)e * HDIM * DDIM,
                                         ub + (size_t)e * HDIM * DDIM,
                                         lists, counts, act, e);
        dim3 g2(T_TOK / 64, DDIM / 256);  // 128 x 4
        k_down<<<g2, 256, 0, stream>>>(act,
                                       db + (size_t)e * DDIM * HDIM,
                                       lists, counts, combine, out, e);
    }
}
